// TransformerStyleGraphAttention_16260746183292
// MI455X (gfx1250) — matrix operand order repaired and checked by reading, not compiled, not run on hardware
//
#include <hip/hip_runtime.h>

// ---------------------------------------------------------------------------
// TransformerStyleGraphAttention for MI455X (gfx1250, wave32)
// B=2, N=8192, M=32, D=256, H=8, HD=32
// All GEMMs via v_wmma_f32_16x16x32_bf16 (fp32 accumulate).
// Round 2: B-fragments hoisted out of the row-tile loop (read once per WG),
//          2 nodes per attention workgroup to amortize L2 weight traffic.
// ---------------------------------------------------------------------------

typedef __bf16 bf16_t;
typedef __attribute__((ext_vector_type(16))) __bf16 v16bf;
typedef __attribute__((ext_vector_type(8)))  float  v8f;

#define BN     16384      // B*N
#define DD     256        // D
#define MM     32         // neighbors
#define HH     8          // heads
#define NPW    2          // nodes per attention workgroup
#define MT     4          // row tiles per attention WG (NPW*32/16)

__device__ __forceinline__ v8f wmma_bf16(v16bf a, v16bf b, v8f c) {
  // 8 args: (neg_a, A, neg_b, B, c_mod, C, reuse_a, reuse_b)
  return __builtin_amdgcn_wmma_f32_16x16x32_bf16(false, a, false, b,
                                                 (short)0, c, false, false);
}

// ---------------------------------------------------------------------------
// Fragment layouts (CDNA5 ISA 7.12.2, wave32):
//  A (16Mx32K bf16): lane = ((k>>3)&1)*16 + m ; elem e = (k>>4)*8 + (k&7)
//  B (32Kx16N bf16): lane = ((k>>3)&1)*16 + n ; elem e = (k>>4)*8 + (k&7)
//  C/D (16x16 f32) : lane = (m>>3)*16 + n    ; vgpr e = m & 7
// Weight fragments stored pre-swizzled: each lane loads 16 contiguous bf16.
// ---------------------------------------------------------------------------

// ---- Kernel 0: fp32 weights -> bf16 B-fragment layout in workspace --------
__global__ __launch_bounds__(256) void prep_weights_kernel(
    const float* __restrict__ Wq, const float* __restrict__ Wk,
    const float* __restrict__ Wv, const float* __restrict__ Wo,
    const float* __restrict__ W1, const float* __restrict__ W2,
    bf16_t* __restrict__ ws) {
  int t = blockIdx.x * 256 + threadIdx.x;
  const float* src;
  bf16_t* dst;
  int ID, i;
  if (t < 262144) {                 // Wq, Wk, Wv, Wo : 256x256 each
    int w = t >> 16; i = t & 65535;
    src = (w == 0) ? Wq : (w == 1) ? Wk : (w == 2) ? Wv : Wo;
    dst = ws + (size_t)w * 65536;
    ID = 256;
  } else if (t < 524288) {          // W1 : 1024x256
    i = t - 262144; src = W1; dst = ws + 262144; ID = 256;
  } else if (t < 786432) {          // W2 : 256x1024
    i = t - 524288; src = W2; dst = ws + 524288; ID = 1024;
  } else return;
  int j = i / ID, d = i % ID;       // out-row j, in-col d ; B[k=d][n=j]=W[j][d]
  int KT = ID >> 5;
  int nt = j >> 4, kt = d >> 5;
  int lane = (((d >> 3) & 1) << 4) | (j & 15);
  int e = (((d >> 4) & 1) << 3) | (d & 7);
  dst[(size_t)((nt * KT + kt) * 32 + lane) * 16 + e] = (bf16_t)src[(size_t)j * ID + d];
}

// ---- Kernel 1: fused QKV projection + masked softmax + attn*V -------------
// 1 workgroup (8 waves) per 2 nodes.  Dynamic LDS = 112640 B.
__global__ __launch_bounds__(256) void attn_kernel(
    const float* __restrict__ nodef, const float* __restrict__ neigh,
    const int* __restrict__ mask,
    const float* __restrict__ bq, const float* __restrict__ bk,
    const float* __restrict__ bv,
    const bf16_t* __restrict__ wqB, const bf16_t* __restrict__ wkB,
    const bf16_t* __restrict__ wvB,
    float* __restrict__ attOut) {
  extern __shared__ char smem[];
  bf16_t* sA   = (bf16_t*)(smem);             // 32 KB : 32 A-frags (64 rows)
  bf16_t* sQA  = (bf16_t*)(smem + 32768);     //  8 KB : 8 A-frags (2 node rows)
  bf16_t* sK   = (bf16_t*)(smem + 40960);     // 32 KB : K[64][256] bf16
  bf16_t* sV   = (bf16_t*)(smem + 73728);     // 32 KB : V[64][256] bf16
  float*  sXst = (float*) (smem + 40960);     // 64 KB staging (overlays sK|sV)
  float*  sNode= (float*) (smem + 106496);    //  2 KB : 2 node rows
  float*  sQ   = (float*) (smem + 108544);    //  2 KB : Q[2][256]
  float*  sP   = (float*) (smem + 110592);    //  2 KB : probs [2][8][32]

  const int tid  = threadIdx.x;
  const int wv   = tid >> 5, lane = tid & 31;
  const int n    = lane & 15, mb = (lane >> 4) << 3;   // C-frag decode
  const size_t node0 = (size_t)blockIdx.x * NPW;

  // --- stage 2 neighbor tiles (64x256 fp32, coalesced) + 2 node rows ---
  const float4* s4 = (const float4*)(neigh + node0 * (MM * DD));
  float4* d4 = (float4*)sXst;
  for (int i = tid; i < 4096; i += 256) d4[i] = s4[i];
  sNode[tid]       = nodef[node0 * DD + tid];
  sNode[tid + 256] = nodef[node0 * DD + 256 + tid];
  __syncthreads();

  // --- build bf16 A-fragments (MT*8 = 32 frags) ---
  for (int idx = tid; idx < MT * 8 * 512; idx += 256) {
    int f = idx >> 9, l = (idx >> 4) & 31, e = idx & 15;
    int mt = f >> 3, kt = f & 7;
    int m = l & 15, kh = l >> 4;
    int k = ((e >> 3) << 4) | (kh << 3) | (e & 7);
    sA[idx] = (bf16_t)sXst[(mt * 16 + m) * DD + kt * 32 + k];
  }
  for (int idx = tid; idx < 4096; idx += 256) {  // Q frags, rows 0..NPW-1 live
    int kt = idx >> 9, l = (idx >> 4) & 31, e = idx & 15;
    int m = l & 15, kh = l >> 4;
    int k = ((e >> 3) << 4) | (kh << 3) | (e & 7);
    sQA[idx] = (m < NPW) ? (bf16_t)sNode[m * DD + kt * 32 + k] : (bf16_t)0.0f;
  }
  __syncthreads();

  // --- K & V projections: wave owns 2 column tiles; each B-fragment is
  //     loaded from L2 exactly once and reused across all MT row tiles. ---
  for (int ntI = 0; ntI < 2; ++ntI) {
    int nt = wv * 2 + ntI;
    v8f accK[MT] = {};
    v8f accV[MT] = {};
#pragma unroll
    for (int kt = 0; kt < 8; ++kt) {
      v16bf fk = *(const v16bf*)&wkB[((nt * 8 + kt) * 32 + lane) * 16];
      v16bf fv = *(const v16bf*)&wvB[((nt * 8 + kt) * 32 + lane) * 16];
#pragma unroll
      for (int mt = 0; mt < MT; ++mt) {
        v16bf a = *(const v16bf*)&sA[((mt * 8 + kt) * 32 + lane) * 16];
        accK[mt] = wmma_bf16(a, fk, accK[mt]);
        accV[mt] = wmma_bf16(a, fv, accV[mt]);
      }
    }
    float bik = bk[nt * 16 + n], biv = bv[nt * 16 + n];
#pragma unroll
    for (int mt = 0; mt < MT; ++mt)
      for (int e = 0; e < 8; ++e) {
        int row = mt * 16 + mb + e, col = nt * 16 + n;
        sK[row * DD + col] = (bf16_t)(accK[mt][e] + bik);
        sV[row * DD + col] = (bf16_t)(accV[mt][e] + biv);
      }
  }
  // --- Q projection (rows 0..NPW-1 of the tile are live) ---
  for (int ntI = 0; ntI < 2; ++ntI) {
    int nt = wv * 2 + ntI;
    v8f acc = {};
#pragma unroll
    for (int kt = 0; kt < 8; ++kt) {
      v16bf a = *(const v16bf*)&sQA[(kt * 32 + lane) * 16];
      v16bf b = *(const v16bf*)&wqB[((nt * 8 + kt) * 32 + lane) * 16];
      acc = wmma_bf16(a, b, acc);
    }
    if (lane < 16) {                 // row m lives at e=m, lanes 0..15
      float bqv = bq[nt * 16 + lane];
      sQ[0 * DD + nt * 16 + lane] = acc[0] + bqv;
      sQ[1 * DD + nt * 16 + lane] = acc[1] + bqv;
    }
  }
  __syncthreads();

  // --- scores + masked softmax: wave wv == head, lane == neighbor ---
  for (int nd = 0; nd < NPW; ++nd) {
    int h = wv, m = lane, base = h * 32;
    float s = 0.f;
    for (int d2 = 0; d2 < 32; ++d2)
      s += sQ[nd * DD + base + d2] * (float)sK[(nd * 32 + m) * DD + base + d2];
    s *= 0.17677669529663687f;                     // 1/sqrt(32)
    if (mask[(node0 + nd) * MM + m] == 0) s = -1e9f;
    float mx = s;
    for (int off = 16; off > 0; off >>= 1) mx = fmaxf(mx, __shfl_xor(mx, off));
    float ex = __expf(s - mx);
    float sm = ex;
    for (int off = 16; off > 0; off >>= 1) sm += __shfl_xor(sm, off);
    sP[nd * (HH * 32) + h * 32 + m] = ex / sm;
  }
  __syncthreads();

  // --- attended[d] = sum_m p[h][m] * V[m][d] ---
  for (int it = 0; it < NPW; ++it) {
    int o = it * 256 + tid;
    int nd = o >> 8, d = o & 255, h = d >> 5;
    float a = 0.f;
    for (int m2 = 0; m2 < 32; ++m2)
      a += sP[nd * (HH * 32) + h * 32 + m2] * (float)sV[(nd * 32 + m2) * DD + d];
    attOut[(node0 + nd) * DD + d] = a;
  }
}

// ---- Kernel 2: Wo + LN1 + FFN + LN2, 32 rows / workgroup ------------------
// Dynamic LDS = 196608 B (CDNA5 allows up to 320 KB per workgroup).
__global__ __launch_bounds__(256) void tail_kernel(
    const float* __restrict__ nodef, const float* __restrict__ attIn,
    const bf16_t* __restrict__ woB, const float* __restrict__ bo,
    const float* __restrict__ g1, const float* __restrict__ be1,
    const bf16_t* __restrict__ w1B, const float* __restrict__ b1,
    const bf16_t* __restrict__ w2B, const float* __restrict__ b2,
    const float* __restrict__ g2, const float* __restrict__ be2,
    float* __restrict__ outp) {
  extern __shared__ char smem[];
  float*  sX = (float*) (smem);             // 32 KB : x (post-LN1)
  float*  sF = (float*) (smem + 32768);     // 32 KB : staging / pre-LN2
  bf16_t* sH = (bf16_t*)(smem + 65536);     // 64 KB : relu(W1 x) [32][1024]
  bf16_t* sG = (bf16_t*)(smem + 131072);    // 64 KB : A-fragments

  const int tid = threadIdx.x, wv = tid >> 5, lane = tid & 31;
  const int n = lane & 15, mb = (lane >> 4) << 3;
  const size_t r0 = (size_t)blockIdx.x * 32;

  // --- stage attended rows ---
  const float4* s4 = (const float4*)(attIn + r0 * DD);
  float4* d4 = (float4*)sF;
  for (int i = tid; i < 2048; i += 256) d4[i] = s4[i];
  __syncthreads();

  // --- A-frags from attended ---
  for (int idx = tid; idx < 8192; idx += 256) {
    int f = idx >> 9, l = (idx >> 4) & 31, e = idx & 15;
    int mt = f >> 3, kt = f & 7;
    int m = l & 15, kh = l >> 4;
    int k = ((e >> 3) << 4) | (kh << 3) | (e & 7);
    sG[idx] = (bf16_t)sF[(mt * 16 + m) * DD + kt * 32 + k];
  }
  __syncthreads();

  // --- Wo projection -> sX (wave owns 2 column tiles, B-frag read once) ---
  for (int ntI = 0; ntI < 2; ++ntI) {
    int nt = wv * 2 + ntI;
    v8f acc[2] = {};
#pragma unroll
    for (int kt = 0; kt < 8; ++kt) {
      v16bf b = *(const v16bf*)&woB[((nt * 8 + kt) * 32 + lane) * 16];
#pragma unroll
      for (int mt = 0; mt < 2; ++mt) {
        v16bf a = *(const v16bf*)&sG[((mt * 8 + kt) * 32 + lane) * 16];
        acc[mt] = wmma_bf16(a, b, acc[mt]);
      }
    }
    float bias = bo[nt * 16 + n];
#pragma unroll
    for (int mt = 0; mt < 2; ++mt)
      for (int e = 0; e < 8; ++e)
        sX[(mt * 16 + mb + e) * DD + nt * 16 + n] = acc[mt][e] + bias;
  }
  __syncthreads();

  // --- residual + LN1 (wave handles 4 rows, fp32) ---
  for (int rr = 0; rr < 4; ++rr) {
    int row = wv * 4 + rr;
    float v[8], sum = 0.f;
    for (int j = 0; j < 8; ++j) {
      int c = lane + j * 32;
      v[j] = nodef[(r0 + row) * DD + c] + sX[row * DD + c];
      sum += v[j];
    }
    for (int off = 16; off > 0; off >>= 1) sum += __shfl_xor(sum, off);
    float mu = sum * (1.f / 256.f);
    float s2 = 0.f;
    for (int j = 0; j < 8; ++j) { float d = v[j] - mu; s2 += d * d; }
    for (int off = 16; off > 0; off >>= 1) s2 += __shfl_xor(s2, off);
    float rs = rsqrtf(s2 * (1.f / 256.f) + 1e-5f);
    for (int j = 0; j < 8; ++j) {
      int c = lane + j * 32;
      sX[row * DD + c] = (v[j] - mu) * rs * g1[c] + be1[c];
    }
  }
  __syncthreads();

  // --- A-frags from x ---
  for (int idx = tid; idx < 8192; idx += 256) {
    int f = idx >> 9, l = (idx >> 4) & 31, e = idx & 15;
    int mt = f >> 3, kt = f & 7;
    int m = l & 15, kh = l >> 4;
    int k = ((e >> 3) << 4) | (kh << 3) | (e & 7);
    sG[idx] = (bf16_t)sX[(mt * 16 + m) * DD + kt * 32 + k];
  }
  __syncthreads();

  // --- W1 + bias + ReLU -> sH (wave owns 8 column tiles) ---
  for (int ntI = 0; ntI < 8; ++ntI) {
    int nt = wv * 8 + ntI;
    v8f acc[2] = {};
#pragma unroll
    for (int kt = 0; kt < 8; ++kt) {
      v16bf b = *(const v16bf*)&w1B[((nt * 8 + kt) * 32 + lane) * 16];
#pragma unroll
      for (int mt = 0; mt < 2; ++mt) {
        v16bf a = *(const v16bf*)&sG[((mt * 8 + kt) * 32 + lane) * 16];
        acc[mt] = wmma_bf16(a, b, acc[mt]);
      }
    }
    float bias = b1[nt * 16 + n];
#pragma unroll
    for (int mt = 0; mt < 2; ++mt)
      for (int e = 0; e < 8; ++e) {
        float hv = acc[mt][e] + bias;
        sH[(mt * 16 + mb + e) * 1024 + nt * 16 + n] = (bf16_t)fmaxf(hv, 0.f);
      }
  }
  __syncthreads();

  // --- A-frags from h (K = 1024 -> 64 fragments) ---
  for (int idx = tid; idx < 32768; idx += 256) {
    int f = idx >> 9, l = (idx >> 4) & 31, e = idx & 15;
    int mt = f >> 5, kt = f & 31;
    int m = l & 15, kh = l >> 4;
    int k = ((e >> 3) << 4) | (kh << 3) | (e & 7);
    sG[idx] = sH[(mt * 16 + m) * 1024 + kt * 32 + k];
  }
  __syncthreads();

  // --- W2 -> sF = x + ffn (wave owns 2 column tiles, 32 k-steps) ---
  for (int ntI = 0; ntI < 2; ++ntI) {
    int nt = wv * 2 + ntI;
    v8f acc[2] = {};
#pragma unroll
    for (int kt = 0; kt < 32; ++kt) {
      v16bf b = *(const v16bf*)&w2B[((nt * 32 + kt) * 32 + lane) * 16];
#pragma unroll
      for (int mt = 0; mt < 2; ++mt) {
        v16bf a = *(const v16bf*)&sG[((mt * 32 + kt) * 32 + lane) * 16];
        acc[mt] = wmma_bf16(a, b, acc[mt]);
      }
    }
    float bias = b2[nt * 16 + n];
#pragma unroll
    for (int mt = 0; mt < 2; ++mt)
      for (int e = 0; e < 8; ++e) {
        int row = mt * 16 + mb + e, col = nt * 16 + n;
        sF[row * DD + col] = sX[row * DD + col] + acc[mt][e] + bias;
      }
  }
  __syncthreads();

  // --- LN2 -> out ---
  for (int rr = 0; rr < 4; ++rr) {
    int row = wv * 4 + rr;
    float v[8], sum = 0.f;
    for (int j = 0; j < 8; ++j) {
      int c = lane + j * 32;
      v[j] = sF[row * DD + c];
      sum += v[j];
    }
    for (int off = 16; off > 0; off >>= 1) sum += __shfl_xor(sum, off);
    float mu = sum * (1.f / 256.f);
    float s2 = 0.f;
    for (int j = 0; j < 8; ++j) { float d = v[j] - mu; s2 += d * d; }
    for (int off = 16; off > 0; off >>= 1) s2 += __shfl_xor(s2, off);
    float rs = rsqrtf(s2 * (1.f / 256.f) + 1e-5f);
    for (int j = 0; j < 8; ++j) {
      int c = lane + j * 32;
      outp[(r0 + row) * DD + c] = (v[j] - mu) * rs * g2[c] + be2[c];
    }
  }
}

// ---------------------------------------------------------------------------
extern "C" void kernel_launch(void* const* d_in, const int* in_sizes, int n_in,
                              void* d_out, int out_size, void* d_ws, size_t ws_size,
                              hipStream_t stream) {
  const float* nodef = (const float*)d_in[0];
  const float* neigh = (const float*)d_in[1];
  const int*   maskp = (const int*)  d_in[2];
  const float* Wq = (const float*)d_in[3];  const float* bq = (const float*)d_in[4];
  const float* Wk = (const float*)d_in[5];  const float* bk = (const float*)d_in[6];
  const float* Wv = (const float*)d_in[7];  const float* bv = (const float*)d_in[8];
  const float* Wo = (const float*)d_in[9];  const float* bo = (const float*)d_in[10];
  const float* g1 = (const float*)d_in[11]; const float* be1 = (const float*)d_in[12];
  const float* W1 = (const float*)d_in[13]; const float* b1 = (const float*)d_in[14];
  const float* W2 = (const float*)d_in[15]; const float* b2 = (const float*)d_in[16];
  const float* g2 = (const float*)d_in[17]; const float* be2 = (const float*)d_in[18];
  float* out = (float*)d_out;

  bf16_t* ws  = (bf16_t*)d_ws;
  bf16_t* wqB = ws;                 // 65536 elems each for q/k/v/o
  bf16_t* wkB = ws + 65536;
  bf16_t* wvB = ws + 131072;
  bf16_t* woB = ws + 196608;
  bf16_t* w1B = ws + 262144;        // 262144 elems
  bf16_t* w2B = ws + 524288;        // 262144 elems

  // 1) weights -> bf16 fragment layout (786432 elements)
  prep_weights_kernel<<<3072, 256, 0, stream>>>(Wq, Wk, Wv, Wo, W1, W2, ws);

  // 2) fused QKV + masked softmax + attn*V ; attended staged in d_out
  attn_kernel<<<BN / NPW, 256, 112640, stream>>>(nodef, neigh, maskp,
                                                 bq, bk, bv,
                                                 wqB, wkB, wvB, out);

  // 3) Wo + LN1 + FFN + LN2 (in-place on d_out rows, 32 rows per WG)
  tail_kernel<<<BN / 32, 256, 196608, stream>>>(nodef, out, woB, bo,
                                                g1, be1, w1B, b1, w2B, b2,
                                                g2, be2, out);
}
